// Gemma4VisionPatchEmbedder_60979945669123
// MI455X (gfx1250) — matrix operand order repaired and checked by reading, compiled, not run on hardware
//
#include <hip/hip_runtime.h>

// Gemma vision patch embedder, fused:
//   out[b,t,h] = sum_k (2*pix[b,t,k]-1) * W[k,h]
//             + mask[b,t] * (pos0[ix[b,t], h] + pos1[iy[b,t], h])
//
// GEMM: M = 32*4096 = 131072, K = 588 (padded to 608), N = 1152.
// bf16 A/B, f32 accumulate via v_wmma_f32_16x16x32_bf16 (wave32).
// v2: ping-pong LDS (1 barrier/iter), next-tile global loads pipelined in
//     registers across WMMAs, register-double-buffered B fragments.

#define KK    588
#define HH    1152
#define MM    (32 * 4096)     // 131072
#define NK    19              // ceil(588/32): K padded to 608

#define LDS_STRIDE 40         // bf16/row: 32 data + 8 pad (80B) -> conflict-free b128

typedef __attribute__((ext_vector_type(16))) __bf16 v16bf;
typedef __attribute__((ext_vector_type(8)))  __bf16 v8bf;
typedef __attribute__((ext_vector_type(8)))  float  v8f;

union Frag { v16bf v; v8bf h[2]; };

__global__ __launch_bounds__(256)
void patch_embed_wmma(const float* __restrict__ pix,      // (M, 588)
                      const int*   __restrict__ ids,      // (M, 2)
                      const float* __restrict__ Wm,       // (588, 1152)
                      const float* __restrict__ pos_tab,  // (2, 1024, 1152)
                      float*       __restrict__ out)      // (M, 1152)
{
    __shared__ __align__(16) __bf16 sA[2][128 * LDS_STRIDE];  // [m][k]
    __shared__ __align__(16) __bf16 sB[2][128 * LDS_STRIDE];  // transposed [n][k]
    __shared__ int   sPX[128];
    __shared__ int   sPY[128];
    __shared__ float sMask[128];

    const int tid  = threadIdx.x;
    const int lane = tid & 31;
    const int wv   = tid >> 5;                 // wave 0..7 -> M sub-strip
    const int mBlock = (int)blockIdx.x / 9;    // N fastest: 9 blocks share an A tile via L2
    const int nBlock = (int)blockIdx.x % 9;
    const int row0 = mBlock * 128;
    const int col0 = nBlock * 128;

    const float* pos0 = pos_tab;               // (1024, 1152)
    const float* pos1 = pos_tab + 1024 * HH;

    // ---- per-row position ids + padding mask ----
    if (tid < 128) {
        int r = row0 + tid;
        int x = ids[2 * r + 0];
        int y = ids[2 * r + 1];
        sMask[tid] = (x == -1 && y == -1) ? 0.0f : 1.0f;
        sPX[tid] = x < 0 ? 0 : x;
        sPY[tid] = y < 0 ? 0 : y;
    }

    // staging roles (fixed per thread)
    const int aR    = tid >> 1;                // A: row within tile
    const int aHalf = tid & 1;                 // A: which 16-k chunk
    const int bN    = tid & 127;               // B: column within tile
    const int bKb   = (tid >> 7) * 16;         // B: which 16-k chunk

    // global load of one K-tile into registers (no LDS yet)
    auto stage_load = [&](int k0, float aV[16], float bV[16]) {
        const float* srcA = pix + (size_t)(row0 + aR) * KK + (k0 + aHalf * 16);
        #pragma unroll
        for (int i = 0; i < 16; ++i) {
            int k = k0 + aHalf * 16 + i;
            aV[i] = (k < KK) ? srcA[i] : 0.5f;       // 0.5 -> 0 after 2x-1 rescale
        }
        const float* srcB = Wm + (size_t)(k0 + bKb) * HH + (col0 + bN);
        #pragma unroll
        for (int i = 0; i < 16; ++i) {
            int k = k0 + bKb + i;
            bV[i] = (k < KK) ? srcB[(size_t)i * HH] : 0.0f;
        }
    };

    // rescale + cvt bf16 + LDS store into buffer `buf`
    auto stage_store = [&](int buf, const float aV[16], const float bV[16]) {
        v8bf alo, ahi, blo, bhi;
        #pragma unroll
        for (int i = 0; i < 8; ++i) {
            alo[i] = (__bf16)(2.0f * aV[i]     - 1.0f);
            ahi[i] = (__bf16)(2.0f * aV[8 + i] - 1.0f);
            blo[i] = (__bf16)bV[i];
            bhi[i] = (__bf16)bV[8 + i];
        }
        v8bf* da = (v8bf*)&sA[buf][aR * LDS_STRIDE + aHalf * 16];
        da[0] = alo;  da[1] = ahi;
        v8bf* db = (v8bf*)&sB[buf][bN * LDS_STRIDE + bKb];
        db[0] = blo;  db[1] = bhi;
    };

    v8f acc[8] = {};   // 8 N-tiles of 16x16 f32 per wave

    // per-lane fragment addressing (wave32 WMMA 16-bit layouts)
    const int kg  = lane >> 4;
    const int l16 = lane & 15;
    const int aBase0 = kg * 16;                // A frag: K = aBase0..+15 (contiguous)

    // ---- prologue: stage tile 0 ----
    float aV[16], bV[16];
    stage_load(0, aV, bV);
    stage_store(0, aV, bV);
    __syncthreads();

    for (int kt = 0; kt < NK; ++kt) {
        const int cur = kt & 1;

        // issue next tile's global loads; they stay in flight across the WMMAs
        float aN[16], bN16[16];
        if (kt + 1 < NK) stage_load((kt + 1) * 32, aN, bN16);

        // ---- A fragment: lane holds row m = wv*16 + l16 ----
        const __bf16* aP = &sA[cur][(wv * 16 + l16) * LDS_STRIDE];
        Frag af;
        af.h[0] = *(const v8bf*)(aP + aBase0);        // K = aBase0 .. +7
        af.h[1] = *(const v8bf*)(aP + aBase0 + 8);    // K = aBase0+8 .. +7

        // ---- B fragments: register double-buffered across the 8 WMMAs ----
        const __bf16* bP = &sB[cur][l16 * LDS_STRIDE + kg * 16];
        Frag bf0, bf1;
        bf0.h[0] = *(const v8bf*)(bP);
        bf0.h[1] = *(const v8bf*)(bP + 8);
        #pragma unroll
        for (int nt = 0; nt < 8; ++nt) {
            Frag& curb = (nt & 1) ? bf1 : bf0;
            Frag& nxtb = (nt & 1) ? bf0 : bf1;
            if (nt < 7) {
                const __bf16* q = bP + (nt + 1) * 16 * LDS_STRIDE;
                nxtb.h[0] = *(const v8bf*)q;
                nxtb.h[1] = *(const v8bf*)(q + 8);
            }
            acc[nt] = __builtin_amdgcn_wmma_f32_16x16x32_bf16(
                false, af.v, false, curb.v, (short)0, acc[nt], false, false);
        }

        // ---- write next tile into the other LDS buffer; one barrier per iter ----
        if (kt + 1 < NK) {
            stage_store(1 - cur, aN, bN16);
            __syncthreads();
        }
    }

    // ---- epilogue: fuse position-embedding gather + mask, write output ----
    // C/D layout: VGPR j holds (M = j + 8*(lane>>4), N = lane&15) of each 16x16 tile.
    #pragma unroll
    for (int nt = 0; nt < 8; ++nt) {
        const int h = col0 + nt * 16 + l16;
        #pragma unroll
        for (int j = 0; j < 8; ++j) {
            const int mloc = wv * 16 + j + (kg << 3);
            const float p = sMask[mloc] *
                (pos0[(size_t)sPX[mloc] * HH + h] + pos1[(size_t)sPY[mloc] * HH + h]);
            out[(size_t)(row0 + mloc) * HH + h] = acc[nt][j] + p;
        }
    }
}

extern "C" void kernel_launch(void* const* d_in, const int* in_sizes, int n_in,
                              void* d_out, int out_size, void* d_ws, size_t ws_size,
                              hipStream_t stream) {
    (void)in_sizes; (void)n_in; (void)out_size; (void)d_ws; (void)ws_size;
    const float* pix = (const float*)d_in[0];   // (32, 4096, 588) f32
    const int*   ids = (const int*)  d_in[1];   // (32, 4096, 2)   i32
    const float* Wm  = (const float*)d_in[2];   // (588, 1152)     f32
    const float* pt  = (const float*)d_in[3];   // (2, 1024, 1152) f32
    float* out = (float*)d_out;                 // (32, 4096, 1152) f32

    dim3 grid((MM / 128) * (HH / 128));         // 1024 * 9 = 9216 workgroups
    patch_embed_wmma<<<grid, 256, 0, stream>>>(pix, ids, Wm, pt, out);
}
